// SpatialGatingUnit_82317343195868
// MI455X (gfx1250) — matrix operand order repaired and checked by reading, not compiled, not run on hardware
//
#include <hip/hip_runtime.h>
#include <hip/hip_bf16.h>

// ---------------------------------------------------------------------------
// SpatialGatingUnit for MI455X (gfx1250, wave32, WMMA bf16 -> f32 accumulate)
// B=32, L=512, H=1024, H2=512, D_ATTN=64
// All GEMM operands arranged so every WMMA fragment load is contiguous 16B.
// Dominant GEMM (Toeplitz) uses async global->LDS staging (ASYNCcnt path).
// ---------------------------------------------------------------------------

#define Bz   32
#define Lz   512
#define Hz   1024
#define H2z  512
#define Dz   64
#define NROW (Bz * Lz)          // 16384 rows of v

typedef __attribute__((ext_vector_type(16))) __bf16 v16bf;
typedef __attribute__((ext_vector_type(8)))  float  v8f;

// ---- CDNA5 WMMA fragment loaders (wave32, 16x16x32 bf16) -------------------
// A (16x32, MxK): lanes 0-15 -> M=lane, K=0..15; lanes 16-31 -> K=16..31
__device__ inline v16bf load_a_frag(const __bf16* base, int lda, int m0, int k0, int lane) {
  int m   = m0 + (lane & 15);
  int sel = (lane >> 4) & 1;
  const __bf16* p = base + (size_t)m * lda + k0;
  v16bf a;
#pragma unroll
  for (int j = 0; j < 8; ++j) a[j]     = p[sel * 16 + j];       // K group 0
#pragma unroll
  for (int j = 0; j < 8; ++j) a[8 + j] = p[sel * 16 + 8 + j];   // K group 1
  return a;
}

// B (32x16, KxN) where memory holds Bt[n][k] (k contiguous): lanes 0-15 -> K=0..15,
// lanes 16-31 -> K=16..31, per-lane 16 contiguous bf16 -> two 16B loads.
__device__ inline v16bf load_bT_frag(const __bf16* baseT, int ldt, int k0, int n0, int lane) {
  int n   = n0 + (lane & 15);
  int sel = (lane >> 4) & 1;
  const __bf16* p = baseT + (size_t)n * ldt + k0 + sel * 16;
  v16bf b;
#pragma unroll
  for (int j = 0; j < 16; ++j) b[j] = p[j];
  return b;
}

#define WMMA_BF16(a, b, c) \
  __builtin_amdgcn_wmma_f32_16x16x32_bf16(false, (a), false, (b), (short)0, (c), false, false)

// ---- CDNA5 async global->LDS copy (ASYNCcnt) -------------------------------
// lds_off: byte offset within wave's LDS (low 32 bits of generic pointer).
__device__ inline void async_copy_b128(unsigned lds_off, const void* gptr) {
  asm volatile("global_load_async_to_lds_b128 %0, %1, off"
               :: "v"(lds_off), "v"(gptr) : "memory");
}
__device__ inline void wait_async0() {
  asm volatile("s_wait_asynccnt 0x0" ::: "memory");
}

// ---------------------------------------------------------------------------
// Kernel 1: prep — transposed bf16 weights + Toeplitz materialization
//   grid: 1536 x 256 == 98304 + 32768 + 262144 elements, one per thread
// ---------------------------------------------------------------------------
__global__ __launch_bounds__(256) void prep_kernel(
    const float* __restrict__ kqv_w, const float* __restrict__ proj_w,
    const float* __restrict__ w_vec,
    __bf16* __restrict__ kqvwT, __bf16* __restrict__ projwT,
    __bf16* __restrict__ W_bf) {
  int i = blockIdx.x * 256 + threadIdx.x;
  const int N1 = (3 * Dz) * H2z;      // 98304  (kqvwT is [192][512])
  const int N2 = H2z * Dz;            // 32768  (projwT is [512][64])
  if (i < N1) {
    int n = i >> 9, k = i & 511;                  // kqvwT[n][k] = kqv_w[k][n]
    kqvwT[i] = (__bf16)kqv_w[(size_t)k * (3 * Dz) + n];
  } else if (i < N1 + N2) {
    int j = i - N1;
    int c = j >> 6, d = j & 63;                   // projwT[c][d] = proj_w[d][c]
    projwT[j] = (__bf16)proj_w[(size_t)d * H2z + c];
  } else {
    int j = i - (N1 + N2);                        // j < 512*512
    int m = j >> 9, l = j & 511;
    W_bf[j] = (__bf16)w_vec[m - l + (Lz - 1)];    // W[m][l] = w_t[l][m]
  }
}

// ---------------------------------------------------------------------------
// Kernel 2: LayerNorm + bf16 conversion of v.  One 256-thread block per row.
// ---------------------------------------------------------------------------
__global__ __launch_bounds__(256) void ln_kernel(
    const float* __restrict__ x, const float* __restrict__ gamma,
    const float* __restrict__ beta,
    __bf16* __restrict__ v_bf, __bf16* __restrict__ vn_bf) {
  int row = blockIdx.x;                    // 0 .. 16383
  int t   = threadIdx.x;
  const float* vp = x + (size_t)row * Hz + H2z;   // v = x[..., 512:]
  float2 val = ((const float2*)vp)[t];
  float s  = val.x + val.y;
  float sq = val.x * val.x + val.y * val.y;
#pragma unroll
  for (int off = 16; off; off >>= 1) {     // wave32 reduction
    s  += __shfl_xor(s,  off, 32);
    sq += __shfl_xor(sq, off, 32);
  }
  __shared__ float red[16];
  int w = t >> 5, lane = t & 31;
  if (lane == 0) { red[w] = s; red[8 + w] = sq; }
  __syncthreads();
  float ts = 0.f, tq = 0.f;
#pragma unroll
  for (int i = 0; i < 8; ++i) { ts += red[i]; tq += red[8 + i]; }
  float mu  = ts * (1.0f / H2z);
  float var = tq * (1.0f / H2z) - mu * mu;
  float inv = rsqrtf(var + 1e-5f);
  int c = 2 * t;
  size_t o = (size_t)row * H2z + c;
  v_bf[o]      = (__bf16)val.x;
  v_bf[o + 1]  = (__bf16)val.y;
  vn_bf[o]     = (__bf16)((val.x - mu) * inv * gamma[c]     + beta[c]);
  vn_bf[o + 1] = (__bf16)((val.y - mu) * inv * gamma[c + 1] + beta[c + 1]);
}

// ---------------------------------------------------------------------------
// Kernel 3: tiled transpose vn[b][l][c] -> vnT[b][c][l]  (32x32 tiles via LDS)
//   grid: 32 batches * 256 tiles = 8192 blocks
// ---------------------------------------------------------------------------
__global__ __launch_bounds__(256) void transpose_kernel(
    const __bf16* __restrict__ vn, __bf16* __restrict__ vnT) {
  __shared__ __bf16 tile[32][33];
  int b  = blockIdx.x >> 8;
  int t8 = blockIdx.x & 255;
  int tr = t8 >> 4, tc = t8 & 15;          // 32x32 tile coords
  int tid = threadIdx.x;
  int r  = tid >> 3;                       // 0..31
  int c4 = (tid & 7) * 4;                  // 0,4,..,28
  const __bf16* src = vn + (size_t)b * Lz * H2z + (size_t)(tr * 32 + r) * H2z + tc * 32 + c4;
#pragma unroll
  for (int j = 0; j < 4; ++j) tile[r][c4 + j] = src[j];
  __syncthreads();
  __bf16* dst = vnT + (size_t)b * Lz * H2z + (size_t)(tc * 32 + r) * Lz + tr * 32 + c4;
#pragma unroll
  for (int j = 0; j < 4; ++j) dst[j] = tile[c4 + j][r];
}

// ---------------------------------------------------------------------------
// Kernel 4: kqv GEMM  C[16384,192] = v_bf @ kqv_w + bias
//   k,q chunks (n<128) -> qkv_bf rows; val chunk (n>=128) -> valT[b][d][l]
// ---------------------------------------------------------------------------
__global__ __launch_bounds__(256) void kqv_gemm(
    const __bf16* __restrict__ A, const __bf16* __restrict__ BwT,
    const float* __restrict__ bias, __bf16* __restrict__ Cout,
    __bf16* __restrict__ valT) {
  int w = threadIdx.x >> 5, lane = threadIdx.x & 31;
  int tile = blockIdx.x * 8 + w;           // < 1024*12
  int mt = tile / 12, nt = tile % 12;
  v8f acc = {};
#pragma unroll 4
  for (int kt = 0; kt < 16; ++kt) {
    v16bf a = load_a_frag(A, H2z, mt * 16, kt * 32, lane);
    v16bf b = load_bT_frag(BwT, H2z, kt * 32, nt * 16, lane);
    acc = WMMA_BF16(a, b, acc);
  }
  int sel = (lane >> 4) & 1;
  int n = nt * 16 + (lane & 15);
  float bn = bias[n];
  if (nt < 8) {                            // k,q chunks: row-major
#pragma unroll
    for (int i = 0; i < 8; ++i) {
      int m = mt * 16 + i + 8 * sel;
      Cout[(size_t)m * (3 * Dz) + n] = (__bf16)(acc[i] + bn);
    }
  } else {                                 // val chunk: write transposed
    int d = n - 2 * Dz;
#pragma unroll
    for (int i = 0; i < 8; ++i) {
      int m = mt * 16 + i + 8 * sel;       // global row = b*512 + l
      int bb = m >> 9, l = m & 511;
      valT[((size_t)bb * Dz + d) * Lz + l] = (__bf16)(acc[i] + bn);
    }
  }
}

// ---------------------------------------------------------------------------
// Kernel 5: tiny attention, fused per (batch, 16-query-row tile).
// ---------------------------------------------------------------------------
__global__ __launch_bounds__(256) void tiny_attn(
    const __bf16* __restrict__ qkv, const __bf16* __restrict__ valT,
    const __bf16* __restrict__ projwT, const float* __restrict__ projb,
    float* __restrict__ attn_out) {
  __shared__ __bf16 q_s[16 * Dz];          //  2 KB
  __shared__ float  sc_s[16 * Lz];         // 32 KB
  __shared__ __bf16 soft_s[16 * Lz];       // 16 KB
  __shared__ __bf16 head_s[16 * Dz];       //  2 KB
  int b  = blockIdx.x >> 5;                // /32
  int qt = blockIdx.x & 31;
  int tid = threadIdx.x, w = tid >> 5, lane = tid & 31;
  const __bf16* qkvb = qkv + (size_t)b * Lz * (3 * Dz);
  const __bf16* valb = valT + (size_t)b * Dz * Lz;

  for (int i = tid; i < 16 * Dz; i += 256) {
    int m = i >> 6, d = i & 63;
    q_s[i] = qkvb[(size_t)(qt * 16 + m) * (3 * Dz) + Dz + d];
  }
  __syncthreads();

  // pass 1: scores[16,512] = q @ k^T * 0.125 (k rows are contiguous in d)
#pragma unroll
  for (int ii = 0; ii < 4; ++ii) {
    int nt = w * 4 + ii;                   // key-row tile 0..31
    v8f acc = {};
#pragma unroll
    for (int kt = 0; kt < 2; ++kt) {       // K = 64
      v16bf a = load_a_frag(q_s, Dz, 0, kt * 32, lane);
      v16bf bb = load_bT_frag(qkvb, 3 * Dz, kt * 32, nt * 16, lane);
      acc = WMMA_BF16(a, bb, acc);
    }
    int sel = (lane >> 4) & 1, n = nt * 16 + (lane & 15);
#pragma unroll
    for (int i = 0; i < 8; ++i) sc_s[(i + 8 * sel) * Lz + n] = acc[i] * 0.125f;
  }
  __syncthreads();

  // pass 2: softmax along 512; wave w owns rows 2w, 2w+1
#pragma unroll
  for (int rr = 0; rr < 2; ++rr) {
    int r = w * 2 + rr;
    float mx = -1e30f;
    for (int c = lane; c < Lz; c += 32) mx = fmaxf(mx, sc_s[r * Lz + c]);
#pragma unroll
    for (int off = 16; off; off >>= 1) mx = fmaxf(mx, __shfl_xor(mx, off, 32));
    float sum = 0.f;
    for (int c = lane; c < Lz; c += 32) {
      float e = __expf(sc_s[r * Lz + c] - mx);
      sc_s[r * Lz + c] = e;
      sum += e;
    }
#pragma unroll
    for (int off = 16; off; off >>= 1) sum += __shfl_xor(sum, off, 32);
    float inv = 1.0f / sum;
    for (int c = lane; c < Lz; c += 32)
      soft_s[r * Lz + c] = (__bf16)(sc_s[r * Lz + c] * inv);
  }
  __syncthreads();

  // pass 3: head[16,64] = soft @ val (valT is [d][l]: contiguous in l)
  if (w < 4) {
    v8f acc = {};
#pragma unroll 4
    for (int kt = 0; kt < 16; ++kt) {      // K = 512
      v16bf a = load_a_frag(soft_s, Lz, 0, kt * 32, lane);
      v16bf bb = load_bT_frag(valb, Lz, kt * 32, w * 16, lane);
      acc = WMMA_BF16(a, bb, acc);
    }
    int sel = (lane >> 4) & 1, n = w * 16 + (lane & 15);
#pragma unroll
    for (int i = 0; i < 8; ++i) head_s[(i + 8 * sel) * Dz + n] = (__bf16)acc[i];
  }
  __syncthreads();

  // pass 4: attn_out[16,512] = head @ proj_w + proj_b (projwT is [c][d])
#pragma unroll
  for (int ii = 0; ii < 4; ++ii) {
    int nt = w * 4 + ii;
    v8f acc = {};
#pragma unroll
    for (int kt = 0; kt < 2; ++kt) {       // K = 64
      v16bf a = load_a_frag(head_s, Dz, 0, kt * 32, lane);
      v16bf bb = load_bT_frag(projwT, Dz, kt * 32, nt * 16, lane);
      acc = WMMA_BF16(a, bb, acc);
    }
    int sel = (lane >> 4) & 1, n = nt * 16 + (lane & 15);
    float bn = projb[n];
#pragma unroll
    for (int i = 0; i < 8; ++i) {
      int m = qt * 16 + i + 8 * sel;
      attn_out[((size_t)b * Lz + m) * H2z + n] = acc[i] + bn;
    }
  }
}

// ---------------------------------------------------------------------------
// Kernel 6: Toeplitz GEMM (block-tiled, async LDS staging) + fused epilogue
//   block = 128x128 tile of out[b]; 8 waves in 4x2; 2x4 frags each.
//   out[b,m,c] = u[b,m,c] * ( (W @ vn[b])[m,c] + t_bias[m] + attn[b,m,c] )
// ---------------------------------------------------------------------------
__global__ __launch_bounds__(256) void sgu_final(
    const __bf16* __restrict__ W, const __bf16* __restrict__ vnT,
    const float* __restrict__ tbias, const float* __restrict__ attn,
    const float* __restrict__ x, float* __restrict__ out) {
  __shared__ __bf16 As[128 * 32];          // A tile  [m][k]  8 KB
  __shared__ __bf16 Bs[128 * 32];          // Bt tile [n][k]  8 KB
  int tid = threadIdx.x, w = tid >> 5, lane = tid & 31;
  int b  = blockIdx.x >> 4;
  int tb = blockIdx.x & 15;
  int m0 = (tb >> 2) * 128;
  int n0 = (tb & 3) * 128;
  int wm = w >> 1, wn = w & 1;             // 4 x 2 wave grid
  const __bf16* vT = vnT + (size_t)b * Lz * H2z;
  unsigned asb = (unsigned)(uintptr_t)(void*)As;   // LDS byte offsets
  unsigned bsb = (unsigned)(uintptr_t)(void*)Bs;
  int ck0 = tid * 2;                       // this thread's 16B-chunk ids
  int row0 = ck0 >> 2,        off0 = (ck0 & 3) * 8;
  int row1 = (ck0 + 1) >> 2,  off1 = ((ck0 + 1) & 3) * 8;

  v8f acc[2][4] = {};
  for (int k0 = 0; k0 < Lz; k0 += 32) {
    __syncthreads();                       // LDS safe to overwrite
    // async-stage A (W rows) and B (vnT rows), 2 x 16B chunks each per thread
    const __bf16* ga0 = W  + (size_t)(m0 + row0) * Lz + k0 + off0;
    const __bf16* ga1 = W  + (size_t)(m0 + row1) * Lz + k0 + off1;
    const __bf16* gb0 = vT + (size_t)(n0 + row0) * Lz + k0 + off0;
    const __bf16* gb1 = vT + (size_t)(n0 + row1) * Lz + k0 + off1;
    async_copy_b128(asb + (unsigned)(row0 * 32 + off0) * 2, ga0);
    async_copy_b128(asb + (unsigned)(row1 * 32 + off1) * 2, ga1);
    async_copy_b128(bsb + (unsigned)(row0 * 32 + off0) * 2, gb0);
    async_copy_b128(bsb + (unsigned)(row1 * 32 + off1) * 2, gb1);
    if (k0 + 32 < Lz) {                    // prefetch next K slice
      __builtin_prefetch(ga0 + 32, 0, 1);
      __builtin_prefetch(gb0 + 32, 0, 1);
    }
    wait_async0();
    __syncthreads();

    v16bf af[2], bfr[4];
#pragma unroll
    for (int i = 0; i < 2; ++i)
      af[i] = load_a_frag(As, 32, (wm * 2 + i) * 16, 0, lane);
#pragma unroll
    for (int j = 0; j < 4; ++j)
      bfr[j] = load_bT_frag(Bs, 32, 0, (wn * 4 + j) * 16, lane);
#pragma unroll
    for (int i = 0; i < 2; ++i)
#pragma unroll
      for (int j = 0; j < 4; ++j)
        acc[i][j] = WMMA_BF16(af[i], bfr[j], acc[i][j]);
  }

  // fused SGU epilogue
  int sel = (lane >> 4) & 1;
#pragma unroll
  for (int i = 0; i < 2; ++i) {
#pragma unroll
    for (int j = 0; j < 4; ++j) {
      int c = n0 + (wn * 4 + j) * 16 + (lane & 15);
#pragma unroll
      for (int e = 0; e < 8; ++e) {
        int m = m0 + (wm * 2 + i) * 16 + e + 8 * sel;
        size_t row = (size_t)b * Lz + m;
        float v_out = acc[i][j][e] + tbias[m] + attn[row * H2z + c];
        out[row * H2z + c] = x[row * Hz + c] * v_out;   // u = x[..., :512]
      }
    }
  }
}

// ---------------------------------------------------------------------------
// Host-side launch
// ---------------------------------------------------------------------------
extern "C" void kernel_launch(void* const* d_in, const int* in_sizes, int n_in,
                              void* d_out, int out_size, void* d_ws, size_t ws_size,
                              hipStream_t stream) {
  const float* x        = (const float*)d_in[0];
  const float* ln_gamma = (const float*)d_in[1];
  const float* ln_beta  = (const float*)d_in[2];
  const float* w_vector = (const float*)d_in[3];
  const float* t_bias   = (const float*)d_in[4];
  const float* kqv_w    = (const float*)d_in[5];
  const float* kqv_b    = (const float*)d_in[6];
  const float* proj_w   = (const float*)d_in[7];
  const float* proj_b   = (const float*)d_in[8];
  float* out = (float*)d_out;

  // workspace layout (bytes)
  char* ws = (char*)d_ws;
  __bf16* v_bf    = (__bf16*)(ws + 0);            // 16384*512*2 = 16 MiB
  __bf16* vn_bf   = (__bf16*)(ws + 16777216);     // 16 MiB
  __bf16* vnT_bf  = (__bf16*)(ws + 33554432);     // 16 MiB
  __bf16* qkv_bf  = (__bf16*)(ws + 50331648);     // 16384*192*2 = 6 MiB (k,q cols)
  __bf16* valT_bf = (__bf16*)(ws + 56623104);     // 32*64*512*2 = 2 MiB
  float*  attn_f  = (float*) (ws + 58720256);     // 16384*512*4 = 32 MiB
  __bf16* W_bf    = (__bf16*)(ws + 92274688);     // 512*512*2
  __bf16* kqvwT   = (__bf16*)(ws + 92798976);     // 192*512*2
  __bf16* projwT  = (__bf16*)(ws + 92995584);     // 512*64*2

  // 1) transposed weights + Toeplitz materialization (393216 elements)
  prep_kernel<<<1536, 256, 0, stream>>>(kqv_w, proj_w, w_vector,
                                        kqvwT, projwT, W_bf);
  // 2) LayerNorm + bf16 conversion of v (one block per row)
  ln_kernel<<<NROW, 256, 0, stream>>>(x, ln_gamma, ln_beta, v_bf, vn_bf);
  // 3) transpose vn -> vnT
  transpose_kernel<<<Bz * 256, 256, 0, stream>>>(vn_bf, vnT_bf);
  // 4) kqv GEMM: 16384x192x512
  kqv_gemm<<<1536, 256, 0, stream>>>(v_bf, kqvwT, kqv_b, qkv_bf, valT_bf);
  // 5) tiny attention: one block per (batch, 16-row q tile)
  tiny_attn<<<Bz * (Lz / 16), 256, 0, stream>>>(qkv_bf, valT_bf, projwT,
                                                proj_b, attn_f);
  // 6) Toeplitz GEMM + epilogue: 32 batches * 16 (128x128) tiles
  sgu_final<<<Bz * 16, 256, 0, stream>>>(W_bf, vnT_bf, t_bias, attn_f, x, out);
}
